// GraphDSTModel_58402965291287
// MI455X (gfx1250) — matrix operand order repaired and checked by reading, not compiled, not run on hardware
//
#include <hip/hip_runtime.h>
#include <hip/hip_bf16.h>

#define D 128
#define NHEADS 8
#define HDIM 16
#define INV_SCALE 0.25f   // 1/sqrt(16)

typedef __attribute__((ext_vector_type(16))) __bf16 v16bf;
typedef __attribute__((ext_vector_type(8)))  float  v8f;

union Frag16 {            // one WMMA 16-element bf16 fragment = 32 bytes
    v16bf v;
    unsigned short u[16];
    uint4 q[2];
};

__device__ __forceinline__ unsigned short f2bf(float f) {
    unsigned u = __float_as_uint(f);
    unsigned r = u + 0x7FFFu + ((u >> 16) & 1u);   // round-to-nearest-even
    return (unsigned short)(r >> 16);
}

// Monotone float<->uint encoding so unsigned atomicMax implements float max.
__device__ __forceinline__ unsigned fenc(float f) {
    unsigned u = __float_as_uint(f);
    return (u & 0x80000000u) ? ~u : (u | 0x80000000u);
}
__device__ __forceinline__ float fdec(unsigned u) {
    return __uint_as_float((u & 0x80000000u) ? (u & 0x7FFFFFFFu) : ~u);
}

// ---------- elementwise f32 -> bf16 (row-major), 8 elems/thread ----------
__global__ void cvt_bf16(const float* __restrict__ in, __bf16* __restrict__ out, int n) {
    int i = (blockIdx.x * blockDim.x + threadIdx.x) * 8;
    if (i >= n) return;
    float4 a = *(const float4*)(in + i);
    float4 b = *(const float4*)(in + i + 4);
    union { unsigned short u[8]; uint4 q; } p;
    p.u[0] = f2bf(a.x); p.u[1] = f2bf(a.y); p.u[2] = f2bf(a.z); p.u[3] = f2bf(a.w);
    p.u[4] = f2bf(b.x); p.u[5] = f2bf(b.y); p.u[6] = f2bf(b.z); p.u[7] = f2bf(b.w);
    *(uint4*)(out + i) = p.q;
}

// ---------- pack W[128x128] f32 into per-lane B-fragment layout (bf16) ----------
// Layout: Wp[((colTile*4 + kbi)*32 + lane)*16 + e], e contiguous (32B per lane).
// Element = W[(kbi*32 + (lane>=16?16:0) + e)*128 + colTile*16 + (lane&15)].
__global__ void pack_w(const float* __restrict__ W, __bf16* __restrict__ Wp) {
    int idx = blockIdx.x * blockDim.x + threadIdx.x;    // 0..1023
    if (idx >= 1024) return;
    const int lane = idx & 31;
    const int kbi  = (idx >> 5) & 3;
    const int ct   = idx >> 7;
    const int lm   = lane & 15;
    const int kb   = kbi * 32 + ((lane >= 16) ? 16 : 0);
    Frag16 f;
#pragma unroll
    for (int e = 0; e < 16; ++e)
        f.u[e] = f2bf(W[(size_t)(kb + e) * D + ct * 16 + lm]);
    uint4* dst = (uint4*)(Wp + (size_t)idx * 16);
    dst[0] = f.q[0];
    dst[1] = f.q[1];
}

// ---------- Y[rows x 128] = Xbf[rows x 128] @ W + bias, bf16 WMMA ----------
// Block = 256 thr = 8 waves; wave w owns cols [16w,16w+16); block owns 4 row tiles.
// All 4 B fragments preloaded once per wave; per tile, all 4 A fragments are
// loaded into distinct registers (one clause, one wait) before 4 chained WMMAs.
__global__ __launch_bounds__(256)
void gemm_bias_wmma(const __bf16* __restrict__ X, const __bf16* __restrict__ Wp,
                    const float* __restrict__ B, float* __restrict__ Y, int rows) {
    const int lane = threadIdx.x & 31;
    const int wave = threadIdx.x >> 5;
    const int lm   = lane & 15;
    const bool hi  = lane >= 16;
    const int colbase  = wave << 4;
    const int totTiles = rows >> 4;
    const int tile0    = blockIdx.x << 2;

    // Preload B fragments for K=0..127 of this wave's column tile.
    Frag16 bf[4];
    const uint4* bp = (const uint4*)Wp + (size_t)(wave * 128 + lane) * 2;
#pragma unroll
    for (int kbi = 0; kbi < 4; ++kbi) {
        bf[kbi].q[0] = bp[kbi * 64 + 0];   // kbi step = 32 lanes * 32B = 64 uint4
        bf[kbi].q[1] = bp[kbi * 64 + 1];
    }

    // Lane-fixed base pointers; all loop offsets are compile-time constants.
    const __bf16* ar = X + (size_t)(tile0 * 16 + lm) * D + (hi ? 16 : 0);
    float*        yr = Y + (size_t)(tile0 * 16 + (hi ? 8 : 0)) * D + colbase + lm;
    const float bias = B[colbase + lm];

#pragma unroll
    for (int t = 0; t < 4; ++t) {
        if (tile0 + t >= totTiles) break;              // wave-uniform guard

        // Gather all 4 A fragments first: 8 b128 loads in flight, single wait.
        Frag16 af[4];
#pragma unroll
        for (int kbi = 0; kbi < 4; ++kbi) {
            af[kbi].q[0] = *(const uint4*)(ar + t * (16 * D) + kbi * 32);
            af[kbi].q[1] = *(const uint4*)(ar + t * (16 * D) + kbi * 32 + 8);
        }

        v8f acc;
#pragma unroll
        for (int r = 0; r < 8; ++r) acc[r] = bias;
#pragma unroll
        for (int kbi = 0; kbi < 4; ++kbi)
            acc = __builtin_amdgcn_wmma_f32_16x16x32_bf16(
                false, af[kbi].v, false, bf[kbi].v, (short)0, acc, false, false);

#pragma unroll
        for (int r = 0; r < 8; ++r)
            yr[t * (16 * D) + r * D] = acc[r];
    }
}

// ---------- edge phase ----------
__global__ void edge_scores(const float* __restrict__ Qp, const float* __restrict__ Kp,
                            const int* __restrict__ src, const int* __restrict__ dst,
                            float* __restrict__ scores, unsigned* __restrict__ smax,
                            int nEdges) {
    int idx = blockIdx.x * blockDim.x + threadIdx.x;
    if (idx >= nEdges * NHEADS) return;
    const int e = idx >> 3, h = idx & 7;
    const int s = src[e], d = dst[e];
    const float4* q = (const float4*)(Qp + (size_t)d * D + h * HDIM);
    const float4* k = (const float4*)(Kp + (size_t)s * D + h * HDIM);
    float acc = 0.f;
#pragma unroll
    for (int i = 0; i < 4; ++i) {
        float4 qa = q[i], ka = k[i];
        acc += qa.x * ka.x + qa.y * ka.y + qa.z * ka.z + qa.w * ka.w;
    }
    acc *= INV_SCALE;
    scores[idx] = acc;
    atomicMax(&smax[d * NHEADS + h], fenc(acc));
}

__global__ void edge_exp(const int* __restrict__ dst, float* __restrict__ scores,
                         const unsigned* __restrict__ smax, float* __restrict__ ssum,
                         int nEdges) {
    int idx = blockIdx.x * blockDim.x + threadIdx.x;
    if (idx >= nEdges * NHEADS) return;
    const int e = idx >> 3, h = idx & 7;
    const int d = dst[e];
    const float m  = fdec(smax[d * NHEADS + h]);
    const float ex = __expf(scores[idx] - m);
    scores[idx] = ex;
    atomicAdd(&ssum[d * NHEADS + h], ex);
}

__global__ void edge_agg(const int* __restrict__ src, const int* __restrict__ dst,
                         const float* __restrict__ scores, const float* __restrict__ ssum,
                         const float* __restrict__ Vp, float* __restrict__ accum,
                         int nEdges) {
    int idx = blockIdx.x * blockDim.x + threadIdx.x;
    if (idx >= nEdges * NHEADS) return;
    const int e = idx >> 3, h = idx & 7;
    const int s = src[e], d = dst[e];
    const float attn = scores[idx] / (ssum[d * NHEADS + h] + 1e-16f);
    const float* v = Vp + (size_t)s * D + h * HDIM;
    float* o = accum + (size_t)d * D + h * HDIM;
#pragma unroll
    for (int i = 0; i < HDIM; ++i) atomicAdd(&o[i], attn * v[i]);
}

extern "C" void kernel_launch(void* const* d_in, const int* in_sizes, int n_in,
                              void* d_out, int out_size, void* d_ws, size_t ws_size,
                              hipStream_t stream) {
    const float* query = (const float*)d_in[0];
    const float* key   = (const float*)d_in[1];
    const float* value = (const float*)d_in[2];
    const int*   eidx  = (const int*)d_in[3];
    const float* wq = (const float*)d_in[4];
    const float* bq = (const float*)d_in[5];
    const float* wk = (const float*)d_in[6];
    const float* bk = (const float*)d_in[7];
    const float* wv = (const float*)d_in[8];
    const float* bv = (const float*)d_in[9];
    const float* wo = (const float*)d_in[10];
    const float* bo = (const float*)d_in[11];

    const int N = in_sizes[0] / D;        // 50000
    const int E = in_sizes[3] / 2;        // 800000
    const int* src = eidx;
    const int* dst = eidx + E;
    const size_t ND = (size_t)N * D;

    // ---- workspace carve-up ----
    char* w = (char*)d_ws;
    float*    Qp     = (float*)w;            w += ND * 4;
    float*    Kp     = (float*)w;            w += ND * 4;
    float*    Vp     = (float*)w;            w += ND * 4;
    float*    scores = (float*)w;            w += (size_t)E * NHEADS * 4;
    unsigned* smax   = (unsigned*)w;         w += (size_t)N * NHEADS * 4;
    float*    ssum   = (float*)w;            w += (size_t)N * NHEADS * 4;
    float*    accum  = (float*)w;            w += ND * 4;
    __bf16*   qbf    = (__bf16*)w;           w += ND * 2;
    __bf16*   kbf    = (__bf16*)w;           w += ND * 2;
    __bf16*   vbf    = (__bf16*)w;           w += ND * 2;
    __bf16*   abf    = (__bf16*)w;           w += ND * 2;
    __bf16*   wqp    = (__bf16*)w;           w += 16384 * 2;
    __bf16*   wkp    = (__bf16*)w;           w += 16384 * 2;
    __bf16*   wvp    = (__bf16*)w;           w += 16384 * 2;
    __bf16*   wop    = (__bf16*)w;           w += 16384 * 2;

    const int totTiles  = N / 16;                       // 3125
    const int gemmBlks  = (totTiles + 3) / 4;           // 782
    const int cvtBlks   = (int)((ND / 8 + 255) / 256);  // 3125
    const int eh        = E * NHEADS;
    const int eBlks     = (eh + 255) / 256;

    // 0) One-time-per-call format conversion / weight packing (cheap, HBM-rate).
    cvt_bf16<<<cvtBlks, 256, 0, stream>>>(query, qbf, (int)ND);
    cvt_bf16<<<cvtBlks, 256, 0, stream>>>(key,   kbf, (int)ND);
    cvt_bf16<<<cvtBlks, 256, 0, stream>>>(value, vbf, (int)ND);
    pack_w<<<4, 256, 0, stream>>>(wq, wqp);
    pack_w<<<4, 256, 0, stream>>>(wk, wkp);
    pack_w<<<4, 256, 0, stream>>>(wv, wvp);
    pack_w<<<4, 256, 0, stream>>>(wo, wop);

    // 1) Projections (WMMA bf16, f32 accumulate)
    gemm_bias_wmma<<<gemmBlks, 256, 0, stream>>>(qbf, wqp, bq, Qp, N);
    gemm_bias_wmma<<<gemmBlks, 256, 0, stream>>>(kbf, wkp, bk, Kp, N);
    gemm_bias_wmma<<<gemmBlks, 256, 0, stream>>>(vbf, wvp, bv, Vp, N);

    // 2) Zero segment-reduction state (deterministic each call).
    hipMemsetAsync(smax,  0, (size_t)N * NHEADS * 4, stream);
    hipMemsetAsync(ssum,  0, (size_t)N * NHEADS * 4, stream);
    hipMemsetAsync(accum, 0, ND * 4, stream);

    // 3) Edge phase (L2-resident gathers + atomics).
    edge_scores<<<eBlks, 256, 0, stream>>>(Qp, Kp, src, dst, scores, smax, E);
    edge_exp   <<<eBlks, 256, 0, stream>>>(dst, scores, smax, ssum, E);
    edge_agg   <<<eBlks, 256, 0, stream>>>(src, dst, scores, ssum, Vp, accum, E);

    // 4) Output projection into d_out.
    cvt_bf16<<<cvtBlks, 256, 0, stream>>>(accum, abf, (int)ND);
    gemm_bias_wmma<<<gemmBlks, 256, 0, stream>>>(abf, wop, bo, (float*)d_out, N);
}
